// Int8LinearGPU_60524679135538
// MI455X (gfx1250) — matrix operand order repaired and checked by reading, not compiled, not run on hardware
//
#include <hip/hip_runtime.h>

typedef __attribute__((ext_vector_type(16))) _Float16 v16h;
typedef __attribute__((ext_vector_type(8)))  _Float16 v8h;
typedef __attribute__((ext_vector_type(4)))  _Float16 v4h;
typedef __attribute__((ext_vector_type(8)))  float    v8f;

#define K_DIM   4096
#define N_DIM   11008
#define M_TILE  128
#define N_TILE  128
#define K_TILE  32
#define LDS_STR 40   // halves per row: 32 data + 8 pad (80B) to spread LDS banks
#define NKT     (K_DIM / K_TILE)   // 128 k-steps

union FragU { v16h v; v8h h[2]; };

__global__ __launch_bounds__(256, 2)
void int8_linear_wmma_f16(const float* __restrict__ X,      // [M, K] fp32
                          const int*   __restrict__ W,      // [N, K] int8-in-int32
                          const float* __restrict__ scale,  // [N]
                          const float* __restrict__ bias,   // [N]
                          float*       __restrict__ Out,    // [M, N] fp32
                          int M)
{
    __shared__ _Float16 lA[2][M_TILE * LDS_STR];
    __shared__ _Float16 lB[2][N_TILE * LDS_STR];

    const int tid  = threadIdx.x;
    const int lane = tid & 31;
    const int wave = tid >> 5;       // 0..7
    const int wm   = wave >> 2;      // 0..1 : 64 output rows per wave
    const int wn   = wave & 3;       // 0..3 : 32 output cols per wave

    // ---- Grid swizzle for L2 reuse (GROUP_M = 8) ----
    const int num_pid_m = M / M_TILE;        // 64
    const int num_pid_n = N_DIM / N_TILE;    // 86
    const int GROUP_M   = 8;
    int bid        = blockIdx.x;
    int group_size = GROUP_M * num_pid_n;
    int group_id   = bid / group_size;
    int first_m    = group_id * GROUP_M;
    int gsm        = num_pid_m - first_m; gsm = (gsm < GROUP_M) ? gsm : GROUP_M;
    int pid_m      = first_m + (bid % gsm);
    int pid_n      = (bid % group_size) / gsm;

    const int m0 = pid_m * M_TILE;
    const int n0 = pid_n * N_TILE;

    // ---- Global staging pattern: tile = 128 rows x 8 vec4; 1024 vec4 / 256 thr = 4 each
    const int rbase = tid >> 3;      // 0..31, +32 per chunk
    const int kv    = tid & 7;       // vec4 column within the 32-wide k tile

    const float4* __restrict__ Xv = (const float4*)X;
    const int4*   __restrict__ Wv = (const int4*)W;

    // ---- Fragment LDS addressing (per ISA VGPR layouts, wave32) ----
    const int lh     = lane & 15;
    const int ahalf  = (lane >> 4) * 8;    // A: k base 0 or 8 (second half at +16)
    const int bhalf  = (lane >> 4) * 8;    // B: k base 0 or 8 (second half at +16)

    v8f acc[4][2];
    #pragma unroll
    for (int mi = 0; mi < 4; ++mi)
        #pragma unroll
        for (int ni = 0; ni < 2; ++ni)
            acc[mi][ni] = (v8f){0.f, 0.f, 0.f, 0.f, 0.f, 0.f, 0.f, 0.f};

    float4 ax[4];
    int4   wx[4];

    // ---- Prologue: tile 0 -> LDS buffer 0 ----
    #pragma unroll
    for (int i = 0; i < 4; ++i) {
        int row = rbase + 32 * i;
        ax[i] = Xv[((size_t)(m0 + row) * K_DIM) / 4 + kv];
        wx[i] = Wv[((size_t)(n0 + row) * K_DIM) / 4 + kv];
    }
    #pragma unroll
    for (int i = 0; i < 4; ++i) {
        int row = rbase + 32 * i;
        v4h ha, hb;
        ha[0] = (_Float16)ax[i].x; ha[1] = (_Float16)ax[i].y;
        ha[2] = (_Float16)ax[i].z; ha[3] = (_Float16)ax[i].w;
        hb[0] = (_Float16)wx[i].x; hb[1] = (_Float16)wx[i].y;
        hb[2] = (_Float16)wx[i].z; hb[3] = (_Float16)wx[i].w;
        *(v4h*)&lA[0][row * LDS_STR + kv * 4] = ha;
        *(v4h*)&lB[0][row * LDS_STR + kv * 4] = hb;
    }
    __syncthreads();

    // ---- Main K loop, double buffered ----
    for (int kt = 0; kt < NKT; ++kt) {
        const int cur = kt & 1;
        const int nxt = cur ^ 1;
        const bool more = (kt + 1) < NKT;

        if (more) {
            const int kc = (kt + 1) * (K_TILE / 4);  // vec4 column base
            #pragma unroll
            for (int i = 0; i < 4; ++i) {
                int row = rbase + 32 * i;
                ax[i] = Xv[((size_t)(m0 + row) * K_DIM) / 4 + kc + kv];
                wx[i] = Wv[((size_t)(n0 + row) * K_DIM) / 4 + kc + kv];
            }
        }

        // B fragments for this wave's two 16-col tiles
        FragU bf[2];
        #pragma unroll
        for (int ni = 0; ni < 2; ++ni) {
            int r = wn * 32 + ni * 16 + lh;
            bf[ni].h[0] = *(const v8h*)&lB[cur][r * LDS_STR + bhalf];
            bf[ni].h[1] = *(const v8h*)&lB[cur][r * LDS_STR + bhalf + 16];
        }

        #pragma unroll
        for (int mi = 0; mi < 4; ++mi) {
            FragU af;
            int r = wm * 64 + mi * 16 + lh;
            af.h[0] = *(const v8h*)&lA[cur][r * LDS_STR + ahalf];        // k 0..7 / 8..15
            af.h[1] = *(const v8h*)&lA[cur][r * LDS_STR + ahalf + 16];   // k 16..23 / 24..31
            #pragma unroll
            for (int ni = 0; ni < 2; ++ni) {
                acc[mi][ni] = __builtin_amdgcn_wmma_f32_16x16x32_f16(
                    false, af.v, false, bf[ni].v,
                    (short)0, acc[mi][ni], false, false);
            }
        }

        if (more) {
            #pragma unroll
            for (int i = 0; i < 4; ++i) {
                int row = rbase + 32 * i;
                v4h ha, hb;
                ha[0] = (_Float16)ax[i].x; ha[1] = (_Float16)ax[i].y;
                ha[2] = (_Float16)ax[i].z; ha[3] = (_Float16)ax[i].w;
                hb[0] = (_Float16)wx[i].x; hb[1] = (_Float16)wx[i].y;
                hb[2] = (_Float16)wx[i].z; hb[3] = (_Float16)wx[i].w;
                *(v4h*)&lA[nxt][row * LDS_STR + kv * 4] = ha;
                *(v4h*)&lB[nxt][row * LDS_STR + kv * 4] = hb;
            }
        }
        __syncthreads();
    }

    // ---- Epilogue: y = acc * scale[n] + bias[n] ----
    #pragma unroll
    for (int ni = 0; ni < 2; ++ni) {
        const int n  = n0 + wn * 32 + ni * 16 + lh;
        const float sc = scale[n];
        const float bs = bias[n];
        #pragma unroll
        for (int mi = 0; mi < 4; ++mi) {
            const int mbase = m0 + wm * 64 + mi * 16 + (lane >> 4) * 8;
            #pragma unroll
            for (int r = 0; r < 8; ++r) {
                Out[(size_t)(mbase + r) * N_DIM + n] = acc[mi][ni][r] * sc + bs;
            }
        }
    }
}

extern "C" void kernel_launch(void* const* d_in, const int* in_sizes, int n_in,
                              void* d_out, int out_size, void* d_ws, size_t ws_size,
                              hipStream_t stream) {
    const float* x  = (const float*)d_in[0];   // [B,S,K] fp32
    const int*   w  = (const int*)d_in[1];     // [N,K] int8-in-int32
    const float* sc = (const float*)d_in[2];   // [N,1]
    const float* bs = (const float*)d_in[3];   // [N]
    float* out      = (float*)d_out;           // [B,S,N] fp32

    const int M = in_sizes[0] / K_DIM;         // 8192
    const int grid = (M / M_TILE) * (N_DIM / N_TILE);  // 64 * 86 = 5504

    int8_linear_wmma_f16<<<dim3(grid), dim3(256), 0, stream>>>(x, w, sc, bs, out, M);
}
